// SA_net_55070070669897
// MI455X (gfx1250) — matrix operand order repaired and checked by reading, not compiled, not run on hardware
//
#include <hip/hip_runtime.h>
#include <cstdint>

typedef _Float16 h16;
typedef __attribute__((ext_vector_type(16))) _Float16 v16h;
typedef __attribute__((ext_vector_type(8)))  float    v8f;

union Frag { v16h h; unsigned u[8]; };

#define NEG_BIG (-3.402823466e38f)

// ---------------------------------------------------------------------------
// WMMA 16x16 output tile: A is LDS (f16, row-major [rows][KP]), B is global
// pre-transposed weights Wt (f16, [NOUT][KP], KP multiple of 32, zero padded).
// Accumulates over the whole K dimension. Returns f32 accumulator fragment.
// Fragment layouts per CDNA5 ISA 7.12.2 (wave32).
// ---------------------------------------------------------------------------
__device__ __forceinline__ v8f wmma_tile(const h16* __restrict__ A, int KP,
                                         const h16* __restrict__ Wt,
                                         int rt, int ct, int lane) {
    v8f acc = {};
    const int m     = rt * 16 + (lane & 15);
    const int ubase = (lane < 16) ? 0 : 4;   // A: k-offset/2 for this lane half
    const int kb    = (lane < 16) ? 0 : 8;   // B: k base for this lane half
    const int nn    = ct * 16 + (lane & 15);
    for (int kc = 0; kc < KP; kc += 32) {
        Frag fa, fb;
        const unsigned* arow = (const unsigned*)(A + (size_t)m * KP + kc);
#pragma unroll
        for (int w = 0; w < 4; ++w) {
            fa.u[w]     = arow[ubase + w];       // k = base + 2w, 2w+1
            fa.u[w + 4] = arow[ubase + w + 8];   // k = base + 16 + 2w, +1
        }
        const unsigned* brow = (const unsigned*)(Wt + (size_t)nn * KP + kc + kb);
#pragma unroll
        for (int w = 0; w < 8; ++w) fb.u[w] = brow[(w < 4) ? w : w + 4]; // k = kb + 2w (+8 for w>=4), +1
        acc = __builtin_amdgcn_wmma_f32_16x16x32_f16(
            false, fa.h, false, fb.h, (short)0, acc, false, false);
    }
    return acc;
}

// One MLP layer on LDS activations: Out = relu(A @ W + b), f16 result in LDS.
__device__ __forceinline__ void gemm_layer(const h16* __restrict__ A, int KP,
                                           const h16* __restrict__ Wt,
                                           const float* __restrict__ bias,
                                           h16* __restrict__ Out, int NOUT, int ROWS) {
    const int lane = threadIdx.x & 31, wave = threadIdx.x >> 5, nw = blockDim.x >> 5;
    const int RT = ROWS / 16, CT = NOUT / 16;
    for (int t = wave; t < RT * CT; t += nw) {
        const int rt = t / CT, ct = t - rt * CT;
        v8f acc = wmma_tile(A, KP, Wt, rt, ct, lane);
        const int nn = ct * 16 + (lane & 15);
        const float bv = bias[nn];
        const int mb = rt * 16 + ((lane < 16) ? 0 : 8);
#pragma unroll
        for (int r = 0; r < 8; ++r) {
            float v = acc[r] + bv;
            v = v > 0.f ? v : 0.f;
            Out[(size_t)(mb + r) * NOUT + nn] = (h16)v;
        }
    }
}

// Final PointConv layer fused with neighbor max-aggregation.
// Centers own 32 consecutive rows (2 row-tiles). tmask holds per-row-tile
// 16-bit validity. Writes f32 features [center][NOUT].
__device__ __forceinline__ void gemm_max_layer(const h16* __restrict__ A, int KP,
                                               const h16* __restrict__ Wt,
                                               const float* __restrict__ bias,
                                               int NOUT, int CENTERS,
                                               const unsigned* __restrict__ tmask,
                                               float* __restrict__ outBase,
                                               size_t outRowOff) {
    const int lane = threadIdx.x & 31, wave = threadIdx.x >> 5, nw = blockDim.x >> 5;
    for (int ct = wave; ct < NOUT / 16; ct += nw) {
        const int nn = ct * 16 + (lane & 15);
        const float bv = bias[nn];
        for (int c = 0; c < CENTERS; ++c) {
            float cm = NEG_BIG;
            for (int hh = 0; hh < 2; ++hh) {
                const int rt = c * 2 + hh;
                v8f acc = wmma_tile(A, KP, Wt, rt, ct, lane);
                const unsigned msk = tmask[rt];
                const int mloc = (lane < 16) ? 0 : 8;
#pragma unroll
                for (int r = 0; r < 8; ++r) {
                    float v = acc[r] + bv;
                    v = v > 0.f ? v : 0.f;
                    if (!((msk >> (mloc + r)) & 1u)) v = NEG_BIG;
                    cm = fmaxf(cm, v);
                }
            }
            // combine lane L with L+16 (same column, other 8 rows): swap halves
            unsigned o = (unsigned)__builtin_amdgcn_ds_swizzle(
                (int)__float_as_uint(cm), 0x401F); // src = lane ^ 16
            float tot = fmaxf(cm, __uint_as_float(o));
            if (lane < 16)
                outBase[(outRowOff + c) * (size_t)NOUT + nn] = tot;
        }
    }
}

// ---------------------------------------------------------------------------
// Weight convert: W f32 [Kin][Nout] -> Wt f16 [Nout][KP] (zero padded K).
// ---------------------------------------------------------------------------
__global__ void wcvt_kernel(const float* __restrict__ W, h16* __restrict__ Wt,
                            int Kin, int Nout, int KP) {
    int t = blockIdx.x * blockDim.x + threadIdx.x;
    if (t >= Nout * KP) return;
    int nn = t / KP, k = t - nn * KP;
    Wt[t] = (h16)((k < Kin) ? W[(size_t)k * Nout + nn] : 0.f);
}

// ---------------------------------------------------------------------------
// Farthest point sampling: one workgroup per cloud, LDS distance array,
// sequential S-step scan with lowest-index-tie argmax (matches jnp.argmax).
// ---------------------------------------------------------------------------
__global__ __launch_bounds__(256) void fps_kernel(const float* __restrict__ pos,
                                                  float* __restrict__ pos_s,
                                                  int n, int S) {
    __shared__ float d[4096];
    __shared__ float rv[256];
    __shared__ int   ri[256];
    __shared__ int   lastSh;
    const int b = blockIdx.x, tid = threadIdx.x, nt = blockDim.x;
    const float* P = pos + (size_t)b * n * 3;
    float* PS = pos_s + (size_t)b * S * 3;
    for (int i = tid; i < n; i += nt) d[i] = 3.402823466e38f;
    if (tid == 0) lastSh = 0;
    __syncthreads();
    for (int s = 0; s < S; ++s) {
        const int last = lastSh;
        const float lx = P[last * 3 + 0], ly = P[last * 3 + 1], lz = P[last * 3 + 2];
        if (tid == 0) { PS[s * 3 + 0] = lx; PS[s * 3 + 1] = ly; PS[s * 3 + 2] = lz; }
        float bv = -1.f; int bi = 0;
        for (int i = tid; i < n; i += nt) {
            const float dx = P[i * 3 + 0] - lx;
            const float dy = P[i * 3 + 1] - ly;
            const float dz = P[i * 3 + 2] - lz;
            float nd = fminf(d[i], dx * dx + dy * dy + dz * dz);
            d[i] = nd;
            if (nd > bv) { bv = nd; bi = i; }
        }
        rv[tid] = bv; ri[tid] = bi;
        __syncthreads();
        for (int off = nt >> 1; off > 0; off >>= 1) {
            if (tid < off) {
                float ov = rv[tid + off]; int oi = ri[tid + off];
                if (ov > rv[tid] || (ov == rv[tid] && oi < ri[tid])) {
                    rv[tid] = ov; ri[tid] = oi;
                }
            }
            __syncthreads();
        }
        if (tid == 0) lastSh = ri[0];
        __syncthreads();
    }
}

// ---------------------------------------------------------------------------
// Ball query: thread per center, streaming K(=32)-nearest within radius.
// Invalid slots marked -1. Earlier index wins ties (matches top_k).
// ---------------------------------------------------------------------------
__global__ void ballq_kernel(const float* __restrict__ pos,
                             const float* __restrict__ posS,
                             int* __restrict__ nbr,
                             int n, int S, float r2, int total) {
    int t = blockIdx.x * blockDim.x + threadIdx.x;
    if (t >= total) return;
    const int b = t / S;
    const float* P = pos + (size_t)b * n * 3;
    const float cx = posS[(size_t)t * 3 + 0];
    const float cy = posS[(size_t)t * 3 + 1];
    const float cz = posS[(size_t)t * 3 + 2];
    float cd[32]; int ci[32];
    int cnt = 0; float mx = 0.f; int mxp = 0;
    for (int i = 0; i < n; ++i) {
        const float dx = P[i * 3 + 0] - cx;
        const float dy = P[i * 3 + 1] - cy;
        const float dz = P[i * 3 + 2] - cz;
        const float d2 = dx * dx + dy * dy + dz * dz;
        if (d2 <= r2) {
            if (cnt < 32) {
                cd[cnt] = d2; ci[cnt] = i; ++cnt;
                if (cnt == 32) {
                    mx = cd[0]; mxp = 0;
                    for (int j = 1; j < 32; ++j) if (cd[j] > mx) { mx = cd[j]; mxp = j; }
                }
            } else if (d2 < mx) {
                cd[mxp] = d2; ci[mxp] = i;
                mx = cd[0]; mxp = 0;
                for (int j = 1; j < 32; ++j) if (cd[j] > mx) { mx = cd[j]; mxp = j; }
            }
        }
    }
    int* out = nbr + (size_t)t * 32;
    for (int j = 0; j < 32; ++j) out[j] = (j < cnt) ? ci[j] : -1;
}

// ---------------------------------------------------------------------------
// Fused PointConv: gather -> MLP(3 layers, WMMA in LDS) -> neighbor max.
// CENTERS centers per workgroup (32 rows each). Output f32 [center][C3].
// ---------------------------------------------------------------------------
template <int CENTERS, int CINP, int C1, int C2, int C3, int XCH, bool FEATPOS>
__global__ void __launch_bounds__(256)
pointconv_kernel(const float* __restrict__ posAll, const float* __restrict__ feat,
                 const float* __restrict__ posS, const int* __restrict__ nbr,
                 const h16* __restrict__ W1t, const float* __restrict__ b1,
                 const h16* __restrict__ W2t, const float* __restrict__ b2,
                 const h16* __restrict__ W3t, const float* __restrict__ b3,
                 float* __restrict__ outFeat, int n, int S) {
    constexpr int ROWS = CENTERS * 32;
    __shared__ __align__(16) h16 inA[ROWS * CINP];
    __shared__ __align__(16) h16 bufX[ROWS * C1];
    __shared__ __align__(16) h16 bufY[ROWS * C2];
    __shared__ unsigned tmask[ROWS / 16];
    const int tid = threadIdx.x;
    for (int i = tid; i < ROWS * CINP; i += blockDim.x) inA[i] = (h16)0.f;
    if (tid < ROWS / 16) tmask[tid] = 0u;
    __syncthreads();
    for (int row = tid; row < ROWS; row += blockDim.x) {
        const int g = blockIdx.x * CENTERS + (row >> 5);
        const int b = g / S;
        const int i = nbr[(size_t)g * 32 + (row & 31)];
        if (i >= 0) {
            atomicOr(&tmask[row >> 4], 1u << (row & 15));
            const float* pc = posS + (size_t)g * 3;
            const float* pj = posAll + ((size_t)b * n + i) * 3;
            h16* dst = inA + (size_t)row * CINP;
            if constexpr (FEATPOS) {
                dst[0] = (h16)pj[0]; dst[1] = (h16)pj[1]; dst[2] = (h16)pj[2];
                dst[3] = (h16)(pj[0] - pc[0]);
                dst[4] = (h16)(pj[1] - pc[1]);
                dst[5] = (h16)(pj[2] - pc[2]);
            } else {
                const float* f = feat + ((size_t)b * n + i) * XCH;
                for (int c = 0; c < XCH; ++c) dst[c] = (h16)f[c];
                dst[XCH + 0] = (h16)(pj[0] - pc[0]);
                dst[XCH + 1] = (h16)(pj[1] - pc[1]);
                dst[XCH + 2] = (h16)(pj[2] - pc[2]);
            }
        }
    }
    __syncthreads();
    gemm_layer(inA, CINP, W1t, b1, bufX, C1, ROWS);
    __syncthreads();
    gemm_layer(bufX, C1, W2t, b2, bufY, C2, ROWS);
    __syncthreads();
    gemm_max_layer(bufY, C2, W3t, b3, C3, CENTERS, tmask,
                   outFeat, (size_t)blockIdx.x * CENTERS);
}

// ---------------------------------------------------------------------------
// Fused mlp3: gather [h2|pos2] -> 259->256->512->512 (relu) -> max over 16
// rows per workgroup, partial maxes to workspace.
// ---------------------------------------------------------------------------
__global__ void __launch_bounds__(256)
mlp3_kernel(const float* __restrict__ h2, const float* __restrict__ pos2,
            const h16* __restrict__ W1t, const float* __restrict__ b1,
            const h16* __restrict__ W2t, const float* __restrict__ b2,
            const h16* __restrict__ W3t, const float* __restrict__ b3,
            float* __restrict__ partials) {
    __shared__ __align__(16) h16 inA[16 * 288];
    __shared__ __align__(16) h16 bufX[16 * 256];
    __shared__ __align__(16) h16 bufY[16 * 512];
    const int tid = threadIdx.x;
    for (int i = tid; i < 16 * 288; i += blockDim.x) inA[i] = (h16)0.f;
    __syncthreads();
    for (int e = tid; e < 16 * 259; e += blockDim.x) {
        const int row = e / 259, c = e - row * 259;
        const size_t gr = (size_t)blockIdx.x * 16 + row;
        const float v = (c < 256) ? h2[gr * 256 + c] : pos2[gr * 3 + (c - 256)];
        inA[row * 288 + c] = (h16)v;
    }
    __syncthreads();
    gemm_layer(inA, 288, W1t, b1, bufX, 256, 16);
    __syncthreads();
    gemm_layer(bufX, 256, W2t, b2, bufY, 512, 16);
    __syncthreads();
    const int lane = tid & 31, wave = tid >> 5, nw = blockDim.x >> 5;
    for (int ct = wave; ct < 512 / 16; ct += nw) {
        v8f acc = wmma_tile(bufY, 512, W3t, 0, ct, lane);
        const int nn = ct * 16 + (lane & 15);
        const float bv = b3[nn];
        float cm = NEG_BIG;
#pragma unroll
        for (int r = 0; r < 8; ++r) {
            float v = acc[r] + bv;
            v = v > 0.f ? v : 0.f;
            cm = fmaxf(cm, v);
        }
        unsigned o = (unsigned)__builtin_amdgcn_ds_swizzle(
            (int)__float_as_uint(cm), 0x401F);
        float tot = fmaxf(cm, __uint_as_float(o));
        if (lane < 16) partials[(size_t)blockIdx.x * 512 + nn] = tot;
    }
}

// Final reduce over 32 partials per batch + write (out, point, b) tuple.
__global__ void finalize_kernel(const float* __restrict__ partials,
                                float* __restrict__ out) {
    const int b = blockIdx.x, c = threadIdx.x;
    float m = NEG_BIG;
    for (int w = 0; w < 32; ++w)
        m = fmaxf(m, partials[((size_t)b * 32 + w) * 512 + c]);
    out[(size_t)b * 512 + c] = m;
    if (b == 0) {
        if (c < 48) out[8192 + c] = 0.f;          // point = zeros [16,3]
        if (c < 16) out[8240 + c] = (float)c;     // b = arange(16)
    }
}

// ---------------------------------------------------------------------------
extern "C" void kernel_launch(void* const* d_in, const int* in_sizes, int n_in,
                              void* d_out, int out_size, void* d_ws, size_t ws_size,
                              hipStream_t stream) {
    (void)in_sizes; (void)n_in; (void)out_size; (void)ws_size;
    const float* x = (const float*)d_in[0];   // [16*4096, 3]
    // d_in[1] = batch (implied by regular layout; unused)
    const float* W[9]; const float* Bv[9];
    for (int i = 0; i < 9; ++i) {
        W[i]  = (const float*)d_in[2 + 2 * i];
        Bv[i] = (const float*)d_in[3 + 2 * i];
    }
    static const int Kin[9]  = {6, 64, 64, 131, 128, 128, 259, 256, 512};
    static const int Nout[9] = {64, 64, 128, 128, 128, 256, 256, 512, 512};
    static const int KP[9]   = {32, 64, 64, 160, 128, 128, 288, 256, 512};

    char* p = (char*)d_ws;
    auto alloc = [&](size_t bytes) -> void* {
        void* r = (void*)p;
        p += (bytes + 255) & ~(size_t)255;
        return r;
    };
    h16* Wt[9];
    for (int i = 0; i < 9; ++i)
        Wt[i] = (h16*)alloc((size_t)Nout[i] * KP[i] * sizeof(h16));
    float* pos1     = (float*)alloc((size_t)16 * 1024 * 3 * 4);
    float* pos2     = (float*)alloc((size_t)16 * 512 * 3 * 4);
    int*   nbr1     = (int*)alloc((size_t)16 * 1024 * 32 * 4);
    int*   nbr2     = (int*)alloc((size_t)16 * 512 * 32 * 4);
    float* h1       = (float*)alloc((size_t)16 * 1024 * 128 * 4);
    float* h2       = (float*)alloc((size_t)16 * 512 * 256 * 4);
    float* partials = (float*)alloc((size_t)512 * 512 * 4);

    for (int i = 0; i < 9; ++i) {
        const int elems = Nout[i] * KP[i];
        wcvt_kernel<<<(elems + 255) / 256, 256, 0, stream>>>(W[i], Wt[i],
                                                             Kin[i], Nout[i], KP[i]);
    }

    // ---- Level 1 ----
    fps_kernel<<<16, 256, 0, stream>>>(x, pos1, 4096, 1024);
    ballq_kernel<<<(16 * 1024 + 255) / 256, 256, 0, stream>>>(
        x, pos1, nbr1, 4096, 1024, 0.04f, 16 * 1024);
    pointconv_kernel<4, 32, 64, 64, 128, 3, true>
        <<<16 * 1024 / 4, 256, 0, stream>>>(
            x, nullptr, pos1, nbr1,
            Wt[0], Bv[0], Wt[1], Bv[1], Wt[2], Bv[2], h1, 4096, 1024);

    // ---- Level 2 ----
    fps_kernel<<<16, 256, 0, stream>>>(pos1, pos2, 1024, 512);
    ballq_kernel<<<(16 * 512 + 255) / 256, 256, 0, stream>>>(
        pos1, pos2, nbr2, 1024, 512, 0.16f, 16 * 512);
    pointconv_kernel<2, 160, 128, 128, 256, 128, false>
        <<<16 * 512 / 2, 256, 0, stream>>>(
            pos1, h1, pos2, nbr2,
            Wt[3], Bv[3], Wt[4], Bv[4], Wt[5], Bv[5], h2, 1024, 512);

    // ---- Global MLP + max pool ----
    mlp3_kernel<<<512, 256, 0, stream>>>(h2, pos2,
                                         Wt[6], Bv[6], Wt[7], Bv[7], Wt[8], Bv[8],
                                         partials);
    finalize_kernel<<<16, 512, 0, stream>>>(partials, (float*)d_out);
}
